// RTE_79611513799034
// MI455X (gfx1250) — matrix operand order repaired and checked by reading, not compiled, not run on hardware
//
#include <hip/hip_runtime.h>
#include <cstdint>
#include <cstddef>

// ---------------------------------------------------------------------------
// RTE forward (masked GRU x2 + attention match-GRU) for MI455X (gfx1250).
// All GEMMs use v_wmma_f32_16x16x32_f16 (wave32, 16x16 tiles, f32 accum).
// Each wave computes a 16x64 strip (4 WMMA tiles): A fragment + all 4 B
// fragments are loaded into distinct registers first (one load clause with
// partial s_wait_loadcnt), then 4 back-to-back WMMAs.  All weights are
// pre-converted to f16 in [N][K] (transposed) layout so every B fragment is
// contiguous along K -> 2x global_load_b128 per fragment.
// ---------------------------------------------------------------------------

typedef _Float16 half_t;
typedef half_t v16h __attribute__((ext_vector_type(16)));
typedef float  v8f  __attribute__((ext_vector_type(8)));

constexpr int BSZ  = 128;   // batch
constexpr int TLEN = 128;   // sequence length
constexpr int EMB  = 300;   // embedding dim
constexpr int EMBP = 320;   // padded to multiple of 32 (WMMA K)
constexpr int ND   = 512;   // hidden dim
constexpr int GG   = 1536;  // 3 * ND (gate dim)
constexpr int NSUB = 4;     // N tiles per wave strip (16x64 output per wave)

__device__ __forceinline__ float sigmoidf_(float x) { return 1.0f / (1.0f + __expf(-x)); }

// ---------------------------------------------------------------------------
// Embedding gather -> f16 (zero-padded K) + sequence mask.
// enc layout: [T][B][EMBP], row r = t*B + b.  tokens layout: [B][T].
// ---------------------------------------------------------------------------
__global__ void embed_kernel(const int* __restrict__ tok, const float* __restrict__ E,
                             half_t* __restrict__ enc, float* __restrict__ mask)
{
    int idx = blockIdx.x * blockDim.x + threadIdx.x;
    int total = TLEN * BSZ * EMBP;
    if (idx >= total) return;
    int k = idx % EMBP;
    int r = idx / EMBP;            // t*B + b
    int t = r / BSZ, b = r % BSZ;
    int tk = tok[b * TLEN + t];
    float v = (k < EMB) ? E[(size_t)tk * EMB + k] : 0.0f;
    enc[idx] = (half_t)v;
    if (k == 0) mask[r] = (tk != 0) ? 1.0f : 0.0f;
}

// f32 [rows][srcK] -> f16 [rows][dstK] with zero K padding (weights stored W[n][k])
__global__ void convert_pad(const float* __restrict__ src, half_t* __restrict__ dst,
                            int rows, int srcK, int dstK)
{
    int idx = blockIdx.x * blockDim.x + threadIdx.x;
    if (idx >= rows * dstK) return;
    int k = idx % dstK, r = idx / dstK;
    dst[idx] = (half_t)((k < srcK) ? src[(size_t)r * srcK + k] : 0.0f);
}

// f32 [K][N] -> f16 [N][K]  (transpose so GEMM B path is contiguous along K)
__global__ void convert_transpose(const float* __restrict__ src, half_t* __restrict__ dst,
                                  int K, int N)
{
    int idx = blockIdx.x * blockDim.x + threadIdx.x;
    if (idx >= N * K) return;
    int k = idx % K, n = idx / K;
    dst[idx] = (half_t)src[(size_t)k * N + n];
}

// ---------------------------------------------------------------------------
// WMMA GEMM: C[M x N] (f32) = A[M x K] * W.T  (+bias / +=C)
//   W is f16 row-major [N x K] (ldb = K): B[k][n] = W[n][k].
//   A_F16 : A stored f16 row-major (lda); else f32 row-major (convert inline).
//   One wave computes a 16 x (16*NSUB) strip.  Per k-chunk: load A fragment +
//   NSUB B fragments into distinct registers, then NSUB back-to-back WMMAs.
//   M mult of 16, N mult of 16*NSUB, K mult of 32.
// ---------------------------------------------------------------------------
template <bool A_F16, bool ACCUM, bool BIAS>
__global__ __launch_bounds__(128)
void wmma_gemm(const void* __restrict__ Aptr, const half_t* __restrict__ W,
               const float* __restrict__ bias, float* __restrict__ C,
               int M, int N, int K, int lda, int ldb)
{
    int wavesPerBlock = blockDim.x >> 5;
    int gwave = blockIdx.x * wavesPerBlock + (threadIdx.x >> 5);
    int stripsN = N / (16 * NSUB);
    int totalStrips = (M >> 4) * stripsN;
    if (gwave >= totalStrips) return;
    int mT = gwave / stripsN, sT = gwave % stripsN;
    int row0 = mT << 4, col0 = sT * (16 * NSUB);

    int lane = threadIdx.x & 31;
    // A fragment addressing (16-bit A 16x32, ISA 7.12.2)
    int am = lane & 15;
    int ak = (lane >> 4) << 3;    // 0 or 8
    // B fragment addressing (16-bit B 32x16)
    int bn = lane & 15;
    int bk = (lane >> 4) << 3;    // 0 or 8
    // C/D addressing (32-bit 16x16, 8 VGPRs)
    int cn = lane & 15;
    int cr = (lane >> 4) << 3;    // 0 or 8

    v8f acc[NSUB];
#pragma unroll
    for (int u = 0; u < NSUB; ++u) {
        if constexpr (ACCUM) {
#pragma unroll
            for (int j = 0; j < 8; ++j)
                acc[u][j] = C[(size_t)(row0 + cr + j) * N + col0 + 16 * u + cn];
        } else if constexpr (BIAS) {
            float bv = bias[col0 + 16 * u + cn];
#pragma unroll
            for (int j = 0; j < 8; ++j) acc[u][j] = bv;
        } else {
#pragma unroll
            for (int j = 0; j < 8; ++j) acc[u][j] = 0.0f;
        }
    }

    const half_t* Ah = (const half_t*)Aptr + (size_t)(row0 + am) * (A_F16 ? lda : 0);
    const float*  Af = (const float*)Aptr + (size_t)(row0 + am) * (A_F16 ? 0 : lda);
    const half_t* Wl = W + (size_t)(col0 + bn) * ldb;   // row (col0+bn) of W

    for (int k0 = 0; k0 < K; k0 += 32) {
        // ---- load phase: A fragment + all NSUB B fragments, distinct regs ----
        v16h a;
#pragma unroll
        for (int j = 0; j < 8; ++j) {
            int kk = k0 + ak + ((j < 4) ? (2 * j) : (16 + 2 * (j - 4)));
            if constexpr (A_F16) {
                a[2 * j]     = Ah[kk];
                a[2 * j + 1] = Ah[kk + 1];
            } else {
                a[2 * j]     = (half_t)Af[kk];
                a[2 * j + 1] = (half_t)Af[kk + 1];
            }
        }
        v16h bfrag[NSUB];
#pragma unroll
        for (int u = 0; u < NSUB; ++u) {
            const half_t* Wu = Wl + (size_t)(16 * u) * ldb;
#pragma unroll
            for (int j = 0; j < 8; ++j) {
                int kk = k0 + bk + ((j < 4) ? (2 * j) : (16 + 2 * (j - 4)));
                bfrag[u][2 * j]     = Wu[kk];
                bfrag[u][2 * j + 1] = Wu[kk + 1];
            }
        }
        // ---- math phase: NSUB back-to-back WMMAs against the shared A ----
#pragma unroll
        for (int u = 0; u < NSUB; ++u)
            acc[u] = __builtin_amdgcn_wmma_f32_16x16x32_f16(
                         false, a, false, bfrag[u], (short)0, acc[u], false, false);
    }

#pragma unroll
    for (int u = 0; u < NSUB; ++u)
#pragma unroll
        for (int j = 0; j < 8; ++j)
            C[(size_t)(row0 + cr + j) * N + col0 + 16 * u + cn] = acc[u][j];
}

// ---------------------------------------------------------------------------
// GRU gate fusion + masked state/output update.
//  GI/GH are [B x 3*ND] with biases already folded in by the GEMMs.
//  hnew = hraw*m + hprev*(1-m)
//  oout = firstStep ? hraw : hraw*m + oprev*(1-m)   (skipped if oout == null)
// ---------------------------------------------------------------------------
__global__ void gru_gates(const float* __restrict__ GI, const float* __restrict__ GH,
                          const float* __restrict__ hprev, const float* __restrict__ mask,
                          const float* __restrict__ oprev, float* __restrict__ hnew,
                          float* __restrict__ oout, int firstStep)
{
    int idx = blockIdx.x * blockDim.x + threadIdx.x;
    if (idx >= BSZ * ND) return;
    int b = idx / ND, j = idx % ND;
    const float* gi = GI + (size_t)b * GG;
    const float* gh = GH + (size_t)b * GG;
    float r = sigmoidf_(gi[j] + gh[j]);
    float z = sigmoidf_(gi[ND + j] + gh[ND + j]);
    float n = tanhf(gi[2 * ND + j] + r * gh[2 * ND + j]);
    float hp = hprev[idx];
    float hraw = (1.0f - z) * n + z * hp;
    float m = mask[b];
    hnew[idx] = hraw * m + hp * (1.0f - m);
    if (oout) {
        float o = firstStep ? hraw : (hraw * m + oprev[idx] * (1.0f - m));
        oout[idx] = o;
    }
}

// scores[b][ty] = W_alpha . tanh(Yw[ty,b,:] + s[b,:])  - 1000*(1-maskP[ty,b])
__global__ void attn_score(const float* __restrict__ Yw, const float* __restrict__ s,
                           const float* __restrict__ Walpha, const float* __restrict__ maskP,
                           float* __restrict__ scores)
{
    int gwave = blockIdx.x * (blockDim.x >> 5) + (threadIdx.x >> 5);
    if (gwave >= BSZ * TLEN) return;
    int b = gwave / TLEN, ty = gwave % TLEN;
    int lane = threadIdx.x & 31;
    const float* yw = Yw + ((size_t)ty * BSZ + b) * ND;
    const float* sb = s + (size_t)b * ND;
    float acc = 0.0f;
    for (int n = lane; n < ND; n += 32)
        acc += tanhf(yw[n] + sb[n]) * Walpha[n];
#pragma unroll
    for (int off = 16; off > 0; off >>= 1)
        acc += __shfl_xor(acc, off, 32);
    if (lane == 0)
        scores[(size_t)b * TLEN + ty] = acc - 1000.0f * (1.0f - maskP[(size_t)ty * BSZ + b]);
}

// softmax over T per batch row, in place
__global__ void attn_softmax(float* __restrict__ scores)
{
    __shared__ float sh[TLEN];
    int b = blockIdx.x;
    int t = threadIdx.x;
    float v = scores[(size_t)b * TLEN + t];
    sh[t] = v; __syncthreads();
    for (int off = TLEN / 2; off > 0; off >>= 1) {
        if (t < off) sh[t] = fmaxf(sh[t], sh[t + off]);
        __syncthreads();
    }
    float mx = sh[0]; __syncthreads();
    float e = __expf(v - mx);
    sh[t] = e; __syncthreads();
    for (int off = TLEN / 2; off > 0; off >>= 1) {
        if (t < off) sh[t] += sh[t + off];
        __syncthreads();
    }
    scores[(size_t)b * TLEN + t] = e / sh[0];
}

// a[b][n] = sum_ty alpha[b][ty] * o_p[ty][b][n]
__global__ void attn_aggregate(const float* __restrict__ alpha, const float* __restrict__ op,
                               float* __restrict__ a)
{
    int b = blockIdx.x;
    for (int n = threadIdx.x; n < ND; n += blockDim.x) {
        float acc = 0.0f;
        for (int ty = 0; ty < TLEN; ++ty)
            acc += alpha[(size_t)b * TLEN + ty] * op[((size_t)ty * BSZ + b) * ND + n];
        a[(size_t)b * ND + n] = acc;
    }
}

// logits = r @ outW.T + outb ; log_softmax over 3 classes
__global__ void logits_kernel(const float* __restrict__ r, const float* __restrict__ outW,
                              const float* __restrict__ outb, float* __restrict__ out)
{
    int b = blockIdx.x;
    int lane = threadIdx.x;   // 32 threads
    float a0 = 0.0f, a1 = 0.0f, a2 = 0.0f;
    const float* rb = r + (size_t)b * ND;
    for (int k = lane; k < ND; k += 32) {
        float rv = rb[k];
        a0 += rv * outW[k];
        a1 += rv * outW[ND + k];
        a2 += rv * outW[2 * ND + k];
    }
#pragma unroll
    for (int off = 16; off > 0; off >>= 1) {
        a0 += __shfl_xor(a0, off, 32);
        a1 += __shfl_xor(a1, off, 32);
        a2 += __shfl_xor(a2, off, 32);
    }
    if (lane == 0) {
        float l0 = a0 + outb[0], l1 = a1 + outb[1], l2 = a2 + outb[2];
        float mx = fmaxf(l0, fmaxf(l1, l2));
        float lse = mx + logf(__expf(l0 - mx) + __expf(l1 - mx) + __expf(l2 - mx));
        out[b * 3 + 0] = l0 - lse;
        out[b * 3 + 1] = l1 - lse;
        out[b * 3 + 2] = l2 - lse;
    }
}

// ---------------------------------------------------------------------------
// Host orchestration
// ---------------------------------------------------------------------------
template <bool AF, bool AC, bool BI>
static void launch_gemm(const void* A, const half_t* W, const float* bias, float* C,
                        int M, int N, int K, int lda, int ldb, hipStream_t s)
{
    int strips = (M / 16) * (N / (16 * NSUB));
    int blocks = (strips + 3) / 4;        // 4 waves (wave32) per 128-thread block
    wmma_gemm<AF, AC, BI><<<blocks, 128, 0, s>>>(A, W, bias, C, M, N, K, lda, ldb);
}

extern "C" void kernel_launch(void* const* d_in, const int* in_sizes, int n_in,
                              void* d_out, int out_size, void* d_ws, size_t ws_size,
                              hipStream_t stream)
{
    (void)in_sizes; (void)n_in; (void)out_size; (void)ws_size;
    const int*   premise    = (const int*)  d_in[0];
    const int*   hypothesis = (const int*)  d_in[1];
    const float* E       = (const float*)d_in[2];
    const float* p_Wih   = (const float*)d_in[3];
    const float* p_Whh   = (const float*)d_in[4];
    const float* p_bih   = (const float*)d_in[5];
    const float* p_bhh   = (const float*)d_in[6];
    const float* h_Wih   = (const float*)d_in[7];
    const float* h_Whh   = (const float*)d_in[8];
    const float* h_bih   = (const float*)d_in[9];
    const float* h_bhh   = (const float*)d_in[10];
    const float* m_Wih   = (const float*)d_in[11];
    const float* m_Whh   = (const float*)d_in[12];
    const float* m_bih   = (const float*)d_in[13];
    const float* m_bhh   = (const float*)d_in[14];
    const float* W_y     = (const float*)d_in[15];
    const float* W_h     = (const float*)d_in[16];
    const float* W_r     = (const float*)d_in[17];
    const float* W_alpha = (const float*)d_in[18];
    const float* out_W   = (const float*)d_in[19];
    const float* out_b   = (const float*)d_in[20];

    char* ws = (char*)d_ws;
    size_t off = 0;
    auto alloc = [&](size_t bytes) -> char* {
        char* p = ws + off;
        off += (bytes + 255) & ~(size_t)255;
        return p;
    };

    half_t* encP = (half_t*)alloc((size_t)TLEN * BSZ * EMBP * 2);
    half_t* encH = (half_t*)alloc((size_t)TLEN * BSZ * EMBP * 2);
    float*  maskP = (float*)alloc((size_t)TLEN * BSZ * 4);
    float*  maskH = (float*)alloc((size_t)TLEN * BSZ * 4);
    half_t* wihP = (half_t*)alloc((size_t)GG * EMBP * 2);
    half_t* wihH = (half_t*)alloc((size_t)GG * EMBP * 2);
    half_t* whhP = (half_t*)alloc((size_t)GG * ND * 2);
    half_t* whhH = (half_t*)alloc((size_t)GG * ND * 2);
    half_t* mWih = (half_t*)alloc((size_t)GG * 2 * ND * 2);
    half_t* mWhh = (half_t*)alloc((size_t)GG * ND * 2);
    half_t* wYt  = (half_t*)alloc((size_t)ND * ND * 2);   // W_y transposed [N][K]
    half_t* wHt  = (half_t*)alloc((size_t)ND * ND * 2);   // W_h transposed
    half_t* wRt  = (half_t*)alloc((size_t)ND * ND * 2);   // W_r transposed
    float*  GI   = (float*)alloc((size_t)TLEN * BSZ * GG * 4);   // 96 MB, reused p/h
    float*  GH   = (float*)alloc((size_t)BSZ * GG * 4);
    float*  GIm  = (float*)alloc((size_t)BSZ * GG * 4);
    float*  GHm  = (float*)alloc((size_t)BSZ * GG * 4);
    float*  hSt0 = (float*)alloc((size_t)BSZ * ND * 4);
    float*  hSt1 = (float*)alloc((size_t)BSZ * ND * 4);
    float*  rSt0 = (float*)alloc((size_t)BSZ * ND * 4);
    float*  rSt1 = (float*)alloc((size_t)BSZ * ND * 4);
    float*  oP   = (float*)alloc((size_t)TLEN * BSZ * ND * 4);
    float*  oH   = (float*)alloc((size_t)TLEN * BSZ * ND * 4);
    float*  Yw   = (float*)alloc((size_t)TLEN * BSZ * ND * 4);
    float*  scores = (float*)alloc((size_t)BSZ * TLEN * 4);
    float*  sbuf = (float*)alloc((size_t)BSZ * ND * 4);
    float*  abuf = (float*)alloc((size_t)BSZ * ND * 4);
    float*  hSt[2] = { hSt0, hSt1 };
    float*  rSt[2] = { rSt0, rSt1 };

    // ---- Phase 0: embeddings, masks, weight conversion (all parallel) ----
    {
        int total = TLEN * BSZ * EMBP;
        embed_kernel<<<(total + 255) / 256, 256, 0, stream>>>(premise, E, encP, maskP);
        embed_kernel<<<(total + 255) / 256, 256, 0, stream>>>(hypothesis, E, encH, maskH);
        auto conv = [&](const float* s, half_t* d, int rows, int sk, int dk) {
            int n = rows * dk;
            convert_pad<<<(n + 255) / 256, 256, 0, stream>>>(s, d, rows, sk, dk);
        };
        conv(p_Wih, wihP, GG, EMB, EMBP);
        conv(h_Wih, wihH, GG, EMB, EMBP);
        conv(p_Whh, whhP, GG, ND, ND);
        conv(h_Whh, whhH, GG, ND, ND);
        conv(m_Wih, mWih, GG, 2 * ND, 2 * ND);
        conv(m_Whh, mWhh, GG, ND, ND);
        auto convT = [&](const float* s, half_t* d) {   // [K][N] -> [N][K]
            int n = ND * ND;
            convert_transpose<<<(n + 255) / 256, 256, 0, stream>>>(s, d, ND, ND);
        };
        convT(W_y, wYt);
        convT(W_h, wHt);
        convT(W_r, wRt);
    }

    const int gateBlocks = (BSZ * ND + 255) / 256;

    // ---- Phase 1: premise GRU ----
    // GI = encP @ p_Wih.T + p_bih  for all timesteps (one big WMMA GEMM)
    launch_gemm<true, false, true>(encP, wihP, p_bih, GI,
                                   TLEN * BSZ, GG, EMBP, EMBP, EMBP, stream);
    hipMemsetAsync(hSt[0], 0, (size_t)BSZ * ND * 4, stream);
    int cur = 0;
    for (int t = 0; t < TLEN; ++t) {
        launch_gemm<false, false, true>(hSt[cur], whhP, p_bhh, GH,
                                        BSZ, GG, ND, ND, ND, stream);
        gru_gates<<<gateBlocks, 256, 0, stream>>>(
            GI + (size_t)t * BSZ * GG, GH, hSt[cur], maskP + (size_t)t * BSZ,
            (t ? oP + (size_t)(t - 1) * BSZ * ND : nullptr),
            hSt[cur ^ 1], oP + (size_t)t * BSZ * ND, t == 0);
        cur ^= 1;
    }

    // ---- Phase 2: hypothesis GRU (h0 = final premise state) ----
    launch_gemm<true, false, true>(encH, wihH, h_bih, GI,
                                   TLEN * BSZ, GG, EMBP, EMBP, EMBP, stream);
    for (int t = 0; t < TLEN; ++t) {
        launch_gemm<false, false, true>(hSt[cur], whhH, h_bhh, GH,
                                        BSZ, GG, ND, ND, ND, stream);
        gru_gates<<<gateBlocks, 256, 0, stream>>>(
            GI + (size_t)t * BSZ * GG, GH, hSt[cur], maskH + (size_t)t * BSZ,
            (t ? oH + (size_t)(t - 1) * BSZ * ND : nullptr),
            hSt[cur ^ 1], oH + (size_t)t * BSZ * ND, t == 0);
        cur ^= 1;
    }

    // ---- Phase 3: Yw = Y @ W_y  (Y rows laid out as (t,b); W_y pre-transposed) ----
    launch_gemm<false, false, false>(oP, wYt, nullptr, Yw,
                                     TLEN * BSZ, ND, ND, ND, ND, stream);

    // ---- Phase 4: attention match-GRU scan over hypothesis outputs ----
    hipMemsetAsync(rSt[0], 0, (size_t)BSZ * ND * 4, stream);
    int rc = 0;
    for (int t = 0; t < TLEN; ++t) {
        const float* ht = oH + (size_t)t * BSZ * ND;
        // s = h_t @ W_h + r @ W_r   (both weights pre-transposed)
        launch_gemm<false, false, false>(ht, wHt, nullptr, sbuf,
                                         BSZ, ND, ND, ND, ND, stream);
        launch_gemm<false, true, false>(rSt[rc], wRt, nullptr, sbuf,
                                        BSZ, ND, ND, ND, ND, stream);
        // alpha = softmax(masked(W_alpha . tanh(Yw + s)))
        attn_score<<<(BSZ * TLEN) / 4, 128, 0, stream>>>(Yw, sbuf, W_alpha, maskP, scores);
        attn_softmax<<<BSZ, TLEN, 0, stream>>>(scores);
        attn_aggregate<<<BSZ, 256, 0, stream>>>(scores, oP, abuf);
        // match GRU: GI = [a, h_t] @ m_Wih.T + m_bih ; GH = r @ m_Whh.T + m_bhh
        launch_gemm<false, false, true>(abuf, mWih, m_bih, GIm,
                                        BSZ, GG, ND, ND, 2 * ND, stream);
        launch_gemm<false, true, false>(ht, mWih + ND, nullptr, GIm,
                                        BSZ, GG, ND, ND, 2 * ND, stream);
        launch_gemm<false, false, true>(rSt[rc], mWhh, m_bhh, GHm,
                                        BSZ, GG, ND, ND, ND, stream);
        gru_gates<<<gateBlocks, 256, 0, stream>>>(
            GIm, GHm, rSt[rc], maskH + (size_t)t * BSZ,
            nullptr, rSt[rc ^ 1], nullptr, 0);
        rc ^= 1;
    }

    // ---- Phase 5: classifier + log_softmax ----
    logits_kernel<<<BSZ, 32, 0, stream>>>(rSt[rc], out_W, out_b, (float*)d_out);
}
